// Processor_76081050682082
// MI455X (gfx1250) — matrix operand order repaired and checked by reading, not compiled, not run on hardware
//
#include <hip/hip_runtime.h>
#include <hip/hip_bf16.h>

#define N_NODES 50000
#define N_EDGES 800000
#define HID     128
#define TPW     5              // 16-row tiles per wave (amortize LDS staging)

typedef __attribute__((ext_vector_type(16))) __bf16 v16bf;
typedef __attribute__((ext_vector_type(8)))  float  v8f;

// ---------------------------------------------------------------------------
// Native CDNA5 f32 max atomic, no return (STOREcnt path), with the uniform
// column offset folded into the instruction's 24-bit immediate offset field.
// ---------------------------------------------------------------------------
template <int BYTEOFF>
__device__ __forceinline__ void atomic_max_f32_imm(float* addr, float val) {
    asm volatile("global_atomic_max_num_f32 %0, %1, off offset:%c2"
                 :: "v"(addr), "v"(val), "n"(BYTEOFF) : "memory");
}

// ---------------------------------------------------------------------------
// Stage a 256x128 f32 weight matrix into LDS as bf16, pre-swizzled into the
// WMMA B-operand lane layout: tiles [kt(0..7)][nt(0..7)], each tile is 32
// lane-vectors of 16 contiguous bf16 (a lane reads 32B = 2x ds_load_b128).
// B layout used here (contiguous k, as in load_a): lane L -> col n = nt*16 + (L&15); kbase = (L>>4)*16;
// slot s -> k = kt*32 + kbase + s.
// ---------------------------------------------------------------------------
__device__ __forceinline__ void stage_B(const float* __restrict__ W,
                                        __bf16* smem, int tid) {
    for (int lv = tid; lv < 2048; lv += 256) {
        const int tile   = lv >> 5;
        const int laneIn = lv & 31;
        const int kt     = tile >> 3;
        const int nt     = tile & 7;
        const int n      = (nt << 4) + (laneIn & 15);
        const int kb     = (kt << 5) + ((laneIn >> 4) << 4);
        __bf16* dst = smem + lv * 16;
#pragma unroll
        for (int s = 0; s < 16; ++s)
            dst[s] = (__bf16)W[(kb + s) * HID + n];
    }
}

__device__ __forceinline__ v16bf loadB(const __bf16* smem, int kt, int nt, int lane) {
    return *(const v16bf*)(smem + (((kt << 3) + nt) * 32 + lane) * 16);
}

// ---------------------------------------------------------------------------
// Load one 16x32 bf16 A-tile slice for this lane from a row pointer.
// A layout used here (contiguous k, same as stage_B): lane L -> row M = L&15; kbase = (L>>4)*16;
// slots 0..15 -> k = k0 + kbase + s
// ---------------------------------------------------------------------------
__device__ __forceinline__ void load_a(const float* __restrict__ rowp, int k0,
                                       int lane, v16bf& a) {
    const int kbase = k0 + ((lane >> 4) << 4);
    const float4* p1 = (const float4*)(rowp + kbase);
    const float4* p2 = (const float4*)(rowp + kbase + 8);
    float4 c0 = p1[0], c1 = p1[1];
    float4 c2 = p2[0], c3 = p2[1];
    a[0]=(__bf16)c0.x; a[1]=(__bf16)c0.y; a[2]=(__bf16)c0.z; a[3]=(__bf16)c0.w;
    a[4]=(__bf16)c1.x; a[5]=(__bf16)c1.y; a[6]=(__bf16)c1.z; a[7]=(__bf16)c1.w;
    a[8]=(__bf16)c2.x; a[9]=(__bf16)c2.y; a[10]=(__bf16)c2.z; a[11]=(__bf16)c2.w;
    a[12]=(__bf16)c3.x; a[13]=(__bf16)c3.y; a[14]=(__bf16)c3.z; a[15]=(__bf16)c3.w;
}

// Same, but map -inf (empty segment) to 0 during conversion.
__device__ __forceinline__ float fixneg(float f) { return f < -1e37f ? 0.0f : f; }
__device__ __forceinline__ void load_a_fix(const float* __restrict__ rowp, int k0,
                                           int lane, v16bf& a) {
    const int kbase = k0 + ((lane >> 4) << 4);
    const float4* p1 = (const float4*)(rowp + kbase);
    const float4* p2 = (const float4*)(rowp + kbase + 8);
    float4 c0 = p1[0], c1 = p1[1];
    float4 c2 = p2[0], c3 = p2[1];
    a[0]=(__bf16)fixneg(c0.x); a[1]=(__bf16)fixneg(c0.y); a[2]=(__bf16)fixneg(c0.z); a[3]=(__bf16)fixneg(c0.w);
    a[4]=(__bf16)fixneg(c1.x); a[5]=(__bf16)fixneg(c1.y); a[6]=(__bf16)fixneg(c1.z); a[7]=(__bf16)fixneg(c1.w);
    a[8]=(__bf16)fixneg(c2.x); a[9]=(__bf16)fixneg(c2.y); a[10]=(__bf16)fixneg(c2.z); a[11]=(__bf16)fixneg(c2.w);
    a[12]=(__bf16)fixneg(c3.x); a[13]=(__bf16)fixneg(c3.y); a[14]=(__bf16)fixneg(c3.z); a[15]=(__bf16)fixneg(c3.w);
}

// ---------------------------------------------------------------------------
// Kernel 0: init segment-max accumulator to -inf
// ---------------------------------------------------------------------------
__global__ __launch_bounds__(256) void init_mbuf(float* __restrict__ mbuf, int n) {
    int i = blockIdx.x * 256 + threadIdx.x;
    if (i < n) mbuf[i] = -__builtin_inff();
}

// ---------------------------------------------------------------------------
// One n-tile of the edge GEMM + scatter-max tail. NT is compile-time so the
// atomic's column offset lands in the instruction immediate.
// ---------------------------------------------------------------------------
template <int NT>
__device__ __forceinline__ void edge_nt(const __bf16* smem, int lane,
                                        const v16bf (&A)[8],
                                        float* const (&colp)[8],
                                        const float (&wrow)[8],
                                        float biasv, float wlv) {
    v8f acc = {};
    v16bf bcur = loadB(smem, 0, NT, lane);
#pragma unroll
    for (int kt = 0; kt < 8; ++kt) {
        v16bf bnext;
        if (kt < 7) bnext = loadB(smem, kt + 1, NT, lane);   // prefetch
        acc = __builtin_amdgcn_wmma_f32_16x16x32_bf16(
            false, A[kt], false, bcur, (short)0, acc, false, false);
        bcur = bnext;
    }
#pragma unroll
    for (int r = 0; r < 8; ++r) {
        const float val = acc[r] + biasv + wrow[r] * wlv;
        atomic_max_f32_imm<NT * 64>(colp[r], val);           // offset = nt*16 floats
    }
}

// ---------------------------------------------------------------------------
// Kernel 1: fused edge GEMM + scatter-max.
// Block = 8 waves; each wave processes TPW consecutive 16-edge tiles.
// ---------------------------------------------------------------------------
__global__ __launch_bounds__(256) void edge_msg_kernel(
    const float* __restrict__ z, const float* __restrict__ weights,
    const int* __restrict__ sources, const int* __restrict__ dests,
    const float* __restrict__ Wm, const float* __restrict__ bm,
    float* __restrict__ mbuf) {
    __shared__ __bf16 smem[32768];   // 64 KB: Wm in WMMA-B swizzled bf16
    const int tid = threadIdx.x;
    stage_B(Wm, smem, tid);

    const int wave = tid >> 5, lane = tid & 31;
    const int lo = lane & 15, hi = lane >> 4;

    // Preload bias and edge-weight-row coefficients (tile-invariant).
    const float* WmLast = Wm + 256 * HID;
    float bias[8], wl[8];
#pragma unroll
    for (int nt = 0; nt < 8; ++nt) {
        const int n = (nt << 4) + lo;
        bias[nt] = bm[n];
        wl[nt]   = WmLast[n];
    }

    __syncthreads();

    const long waveSlot = (long)blockIdx.x * 8 + wave;
    for (int t = 0; t < TPW; ++t) {
        const long e0     = (waveSlot * TPW + t) * 16;
        const int  e_lane = (int)e0 + lo;

        const int   drow   = dests[e_lane];
        const int   srow   = sources[e_lane];
        const float w_lane = weights[e_lane];

        const float* dptr = z + (long)drow * HID;
        const float* sptr = z + (long)srow * HID;

        // Register-resident A: 16 edges x 256 K (dest rows, then src rows).
        v16bf A[8];
#pragma unroll
        for (int kt = 0; kt < 4; ++kt) load_a(dptr, kt * 32, lane, A[kt]);
#pragma unroll
        for (int kt = 0; kt < 4; ++kt) load_a(sptr, kt * 32, lane, A[4 + kt]);

        // nt-invariant scatter state: per-row base pointer (+lane column) and
        // per-row edge weight, fetched via wave32 shuffles.
        float* colp[8];
        float  wrow[8];
#pragma unroll
        for (int r = 0; r < 8; ++r) {
            const int mrow = r + (hi << 3);
            const int dm   = __shfl(drow, mrow);
            wrow[r]        = __shfl(w_lane, mrow);
            colp[r]        = mbuf + (long)dm * HID + lo;
        }

        edge_nt<0>(smem, lane, A, colp, wrow, bias[0], wl[0]);
        edge_nt<1>(smem, lane, A, colp, wrow, bias[1], wl[1]);
        edge_nt<2>(smem, lane, A, colp, wrow, bias[2], wl[2]);
        edge_nt<3>(smem, lane, A, colp, wrow, bias[3], wl[3]);
        edge_nt<4>(smem, lane, A, colp, wrow, bias[4], wl[4]);
        edge_nt<5>(smem, lane, A, colp, wrow, bias[5], wl[5]);
        edge_nt<6>(smem, lane, A, colp, wrow, bias[6], wl[6]);
        edge_nt<7>(smem, lane, A, colp, wrow, bias[7], wl[7]);
    }
}

// ---------------------------------------------------------------------------
// Kernel 2: node update  h = concat(z, fix(m)) @ Wu + bu
// Block = 8 waves; each wave processes TPW consecutive 16-node tiles.
// ---------------------------------------------------------------------------
__global__ __launch_bounds__(256) void node_update_kernel(
    const float* __restrict__ z, const float* __restrict__ mbuf,
    const float* __restrict__ Wu, const float* __restrict__ bu,
    float* __restrict__ h) {
    __shared__ __bf16 smem[32768];
    const int tid = threadIdx.x;
    stage_B(Wu, smem, tid);

    const int wave = tid >> 5, lane = tid & 31;
    const int lo = lane & 15, hi = lane >> 4;

    float bias[8];
#pragma unroll
    for (int nt = 0; nt < 8; ++nt) bias[nt] = bu[(nt << 4) + lo];

    __syncthreads();

    const long waveSlot = (long)blockIdx.x * 8 + wave;
    for (int t = 0; t < TPW; ++t) {
        const long n0 = (waveSlot * TPW + t) * 16;

        int node_lane = (int)n0 + lo;
        node_lane = node_lane < (N_NODES - 1) ? node_lane : (N_NODES - 1);

        const float* zr = z    + (long)node_lane * HID;
        const float* mr = mbuf + (long)node_lane * HID;

        v16bf A[8];
#pragma unroll
        for (int kt = 0; kt < 4; ++kt) load_a(zr, kt * 32, lane, A[kt]);
#pragma unroll
        for (int kt = 0; kt < 4; ++kt) load_a_fix(mr, kt * 32, lane, A[4 + kt]);

        // nt-invariant output pointers (+lane column) and validity flags.
        float* outp[8];
        bool   ok[8];
#pragma unroll
        for (int r = 0; r < 8; ++r) {
            const long node = n0 + r + (hi << 3);
            ok[r]   = node < N_NODES;
            outp[r] = h + node * HID + lo;
        }

#pragma unroll
        for (int nt = 0; nt < 8; ++nt) {
            v8f acc = {};
            v16bf bcur = loadB(smem, 0, nt, lane);
#pragma unroll
            for (int kt = 0; kt < 8; ++kt) {
                v16bf bnext;
                if (kt < 7) bnext = loadB(smem, kt + 1, nt, lane);  // prefetch
                acc = __builtin_amdgcn_wmma_f32_16x16x32_bf16(
                    false, A[kt], false, bcur, (short)0, acc, false, false);
                bcur = bnext;
            }
#pragma unroll
            for (int r = 0; r < 8; ++r) {
                if (ok[r]) outp[r][nt * 16] = acc[r] + bias[nt];  // imm offset
            }
        }
    }
}

// ---------------------------------------------------------------------------
extern "C" void kernel_launch(void* const* d_in, const int* in_sizes, int n_in,
                              void* d_out, int out_size, void* d_ws, size_t ws_size,
                              hipStream_t stream) {
    const float* z       = (const float*)d_in[0];
    const float* weights = (const float*)d_in[1];
    const int*   sources = (const int*)d_in[2];
    const int*   dests   = (const int*)d_in[3];
    const float* Wm      = (const float*)d_in[4];
    const float* bm      = (const float*)d_in[5];
    const float* Wu      = (const float*)d_in[6];
    const float* bu      = (const float*)d_in[7];
    float*       h       = (float*)d_out;
    float*       mbuf    = (float*)d_ws;   // N_NODES * HID f32 = 25.6 MB

    const int melems = N_NODES * HID;
    init_mbuf<<<(melems + 255) / 256, 256, 0, stream>>>(mbuf, melems);

    // Edge tiles: 800000/16 = 50000; per block: 8 waves * TPW(5) tiles.
    // 50000 / 40 = 1250 blocks exactly.
    edge_msg_kernel<<<N_EDGES / (16 * 8 * TPW), 256, 0, stream>>>(
        z, weights, sources, dests, Wm, bm, mbuf);

    // Node tiles: ceil(50000/16) = 3125; wave-slots = ceil(3125/TPW) = 625;
    // blocks = ceil(625/8) = 79 (overhang handled by clamp+guard).
    const int node_tiles  = (N_NODES + 15) / 16;
    const int node_slots  = (node_tiles + TPW - 1) / TPW;
    const int node_blocks = (node_slots + 7) / 8;
    node_update_kernel<<<node_blocks, 256, 0, stream>>>(z, mbuf, Wu, bu, h);
}
